// GraphAttentionLayer_35931696398881
// MI455X (gfx1250) — matrix operand order repaired and checked by reading, not compiled, not run on hardware
//
#include <hip/hip_runtime.h>
#include <hip/hip_bf16.h>

typedef __attribute__((ext_vector_type(16))) _Float16 v16h;
typedef __attribute__((ext_vector_type(8)))  float    v8f;

#define N_NODES 8192
#define F_IN    128
#define F_OUT   64
#define NW      8          // waves per block in the attention kernel
#define LEAKY   0.2f

// ---------------- Kernel 1: h = x @ W + b ; also hT (f16, transposed) ----
__global__ __launch_bounds__(256)
void gat_h_kernel(const float* __restrict__ x, const float* __restrict__ W,
                  const float* __restrict__ bias,
                  float* __restrict__ h, _Float16* __restrict__ hT)
{
    int idx = blockIdx.x * 256 + threadIdx.x;     // [0, 8192*64)
    int row = idx >> 6;
    int col = idx & 63;
    float s = bias[col];
    #pragma unroll 8
    for (int k = 0; k < F_IN; ++k)
        s += x[(long)row * F_IN + k] * W[k * F_OUT + col];
    h[(long)row * F_OUT + col] = s;
    hT[(long)col * N_NODES + row] = (_Float16)s;
}

// ---------------- Kernel 2: f1 = h @ a1, f2 = h @ a2 ---------------------
__global__ __launch_bounds__(256)
void gat_f_kernel(const float* __restrict__ h, const float* __restrict__ a1,
                  const float* __restrict__ a2,
                  float* __restrict__ f1, float* __restrict__ f2)
{
    int row = blockIdx.x * 256 + threadIdx.x;
    const float4* hp = (const float4*)(h + (long)row * F_OUT);
    float s1 = 0.f, s2 = 0.f;
    #pragma unroll
    for (int q = 0; q < F_OUT / 4; ++q) {
        float4 v  = hp[q];
        float4 b1 = ((const float4*)a1)[q];
        float4 b2 = ((const float4*)a2)[q];
        s1 += v.x * b1.x + v.y * b1.y + v.z * b1.z + v.w * b1.w;
        s2 += v.x * b2.x + v.y * b2.y + v.z * b2.z + v.w * b2.w;
    }
    f1[row] = s1;
    f2[row] = s2;
}

// ---------------- Kernel 3: fused masked-softmax + attn@h (flash style) --
// One block = 16 output rows. 8 waves split the j-range (online softmax per
// wave), partials merged in LDS. WMMA: A = 16x32 f16 probability tile,
// B = 32x16 f16 tile of hT, C = f32 accumulator (4 N-tiles -> 64 cols).
__global__ __launch_bounds__(NW * 32)
void gat_attn_kernel(const int* __restrict__ adj,
                     const float* __restrict__ f1v,
                     const float* __restrict__ f2v,
                     const _Float16* __restrict__ hT,
                     float* __restrict__ out)
{
    __shared__ float sAcc[NW][16][F_OUT];   // 32 KB
    __shared__ float sM[NW][16];
    __shared__ float sD[NW][16];

    const int tid  = threadIdx.x;
    const int wave = tid >> 5;
    const int lane = tid & 31;
    const int li   = lane & 15;             // output row within block
    const int hi   = lane >> 4;             // lane half
    const int i0   = blockIdx.x * 16;
    const int row  = i0 + li;

    const float f1i = f1v[row];
    const float NEGINF = -__builtin_inff();

    float m    = NEGINF;                    // running max
    float dsum = 0.0f;                      // running denominator
    v8f c0 = {}, c1 = {}, c2 = {}, c3 = {}; // 16x64 f32 accumulator

    // A-matrix K runs for this lane half: {0..15} or {16..31}
    const int kb0 = hi ? 16 : 0;
    const int kb1 = kb0 + 8;
    const long adjRow = (long)row * N_NODES;

    const int CHUNKS = N_NODES / (NW * 32);
    for (int chunk = 0; chunk < CHUNKS; ++chunk) {
        const int j0 = (chunk * NW + wave) * 32;

        // adjacency: two contiguous runs of 8 ints (whole 128B line / 2 lanes)
        int4 a00 = *(const int4*)(adj + adjRow + j0 + kb0);
        int4 a01 = *(const int4*)(adj + adjRow + j0 + kb0 + 4);
        int4 a10 = *(const int4*)(adj + adjRow + j0 + kb1);
        int4 a11 = *(const int4*)(adj + adjRow + j0 + kb1 + 4);
        // f2: same runs (L2 resident)
        float4 g00 = *(const float4*)(f2v + j0 + kb0);
        float4 g01 = *(const float4*)(f2v + j0 + kb0 + 4);
        float4 g10 = *(const float4*)(f2v + j0 + kb1);
        float4 g11 = *(const float4*)(f2v + j0 + kb1 + 4);

        int   av[16] = {a00.x,a00.y,a00.z,a00.w, a01.x,a01.y,a01.z,a01.w,
                        a10.x,a10.y,a10.z,a10.w, a11.x,a11.y,a11.z,a11.w};
        float fv[16] = {g00.x,g00.y,g00.z,g00.w, g01.x,g01.y,g01.z,g01.w,
                        g10.x,g10.y,g10.z,g10.w, g11.x,g11.y,g11.z,g11.w};

        float z[16];
        float cmax = NEGINF;
        #pragma unroll
        for (int t = 0; t < 16; ++t) {
            float l = f1i + fv[t];
            l = (l > 0.0f) ? l : (LEAKY * l);     // LeakyReLU
            z[t] = l;
            cmax = (av[t] != 0) ? fmaxf(cmax, l) : cmax;
        }
        // combine with partner lane holding same row (other K half)
        cmax = fmaxf(cmax, __shfl_xor(cmax, 16, 32));
        float mnew  = fmaxf(m, cmax);
        float scale = (mnew == m) ? 1.0f : __expf(m - mnew);

        // probabilities (A tile, f16) + partial denominator
        v16h A;
        float psum = 0.0f;
        #pragma unroll
        for (int t = 0; t < 16; ++t) {
            float p = (av[t] != 0) ? __expf(z[t] - mnew) : 0.0f;
            psum += p;
            A[t] = (_Float16)p;
        }
        psum += __shfl_xor(psum, 16, 32);
        dsum  = dsum * scale + psum;
        m     = mnew;

        // rescale accumulator rows: C lane layout row = r + 8*hi
        #pragma unroll
        for (int r = 0; r < 8; ++r) {
            float sr = __shfl(scale, r + 8 * hi, 32);
            c0[r] *= sr; c1[r] *= sr; c2[r] *= sr; c3[r] *= sr;
        }

        // B tiles: lane holds column n = tile*16 + li, K = hi?16..31:0..15
        const _Float16* bbase = hT + j0 + (hi ? 16 : 0);
        v16h B0 = *(const v16h*)(bbase + (long)(li +  0) * N_NODES);
        v16h B1 = *(const v16h*)(bbase + (long)(li + 16) * N_NODES);
        v16h B2 = *(const v16h*)(bbase + (long)(li + 32) * N_NODES);
        v16h B3 = *(const v16h*)(bbase + (long)(li + 48) * N_NODES);

        c0 = __builtin_amdgcn_wmma_f32_16x16x32_f16(false, A, false, B0, (short)0, c0, false, false);
        c1 = __builtin_amdgcn_wmma_f32_16x16x32_f16(false, A, false, B1, (short)0, c1, false, false);
        c2 = __builtin_amdgcn_wmma_f32_16x16x32_f16(false, A, false, B2, (short)0, c2, false, false);
        c3 = __builtin_amdgcn_wmma_f32_16x16x32_f16(false, A, false, B3, (short)0, c3, false, false);
    }

    // publish per-wave partials
    if (lane < 16) { sM[wave][li] = m; sD[wave][li] = dsum; }
    #pragma unroll
    for (int r = 0; r < 8; ++r) {
        int rr = r + 8 * hi;
        sAcc[wave][rr][ 0 + li] = c0[r];
        sAcc[wave][rr][16 + li] = c1[r];
        sAcc[wave][rr][32 + li] = c2[r];
        sAcc[wave][rr][48 + li] = c3[r];
    }
    __syncthreads();

    // flash merge across waves + ELU + store (16x64 outputs, 256 threads)
    for (int e = tid; e < 16 * F_OUT; e += NW * 32) {
        int rr = e >> 6, cc = e & 63;
        float M = NEGINF;
        #pragma unroll
        for (int w = 0; w < NW; ++w) M = fmaxf(M, sM[w][rr]);
        float D = 0.0f, V = 0.0f;
        #pragma unroll
        for (int w = 0; w < NW; ++w) {
            float s = __expf(sM[w][rr] - M);
            D += sD[w][rr] * s;
            V += sAcc[w][rr][cc] * s;
        }
        float o = V / D;
        o = (o > 0.0f) ? o : (__expf(o) - 1.0f);   // ELU
        out[(long)(i0 + rr) * F_OUT + cc] = o;
    }
}

extern "C" void kernel_launch(void* const* d_in, const int* in_sizes, int n_in,
                              void* d_out, int out_size, void* d_ws, size_t ws_size,
                              hipStream_t stream) {
    const float* x    = (const float*)d_in[0];
    const int*   adj  = (const int*)d_in[1];
    const float* W    = (const float*)d_in[2];
    const float* bias = (const float*)d_in[3];
    const float* a1   = (const float*)d_in[4];
    const float* a2   = (const float*)d_in[5];
    float* out = (float*)d_out;

    char* ws = (char*)d_ws;
    float*    h  = (float*)ws;                                   // 2 MB
    _Float16* hT = (_Float16*)(ws + (size_t)2 * 1024 * 1024);    // 1 MB
    float*    f1 = (float*)(ws + (size_t)3 * 1024 * 1024);       // 32 KB
    float*    f2 = (float*)(ws + (size_t)3 * 1024 * 1024 + 64 * 1024);

    gat_h_kernel<<<(N_NODES * F_OUT) / 256, 256, 0, stream>>>(x, W, bias, h, hT);
    gat_f_kernel<<<N_NODES / 256, 256, 0, stream>>>(h, a1, a2, f1, f2);
    gat_attn_kernel<<<N_NODES / 16, NW * 32, 0, stream>>>(adj, f1, f2, hT, out);
}
